// CascadeTransformerMM_12206297055439
// MI455X (gfx1250) — 16-bit and 8-bit matrix operand order repaired and checked by reading, compiled, not run on hardware
//
#include <hip/hip_runtime.h>
#include <hip/hip_bf16.h>


typedef __attribute__((ext_vector_type(8)))  int    v8i;
typedef __attribute__((ext_vector_type(8)))  float  v8f;
typedef __attribute__((ext_vector_type(16))) __bf16 v16bf;

#define BB   8
#define SS   512
#define EE   2048
#define HH   2048
#define NTOK (BB*SS)
#define MT   (NTOK/16)
#define NT   (HH/16)
#define KT8  (EE/64)
#define KT16 (EE/32)

__device__ __forceinline__ unsigned f2bf(float f) {
  unsigned u = __float_as_uint(f);
  return ((u + 0x7fffu + ((u >> 16) & 1u)) >> 16) & 0xffffu;
}
__device__ __forceinline__ float sigm(float x) { return 1.f / (1.f + __expf(-x)); }

__global__ void init_amax_kernel(unsigned* amax) {
  if (threadIdx.x < 3) amax[threadIdx.x] = 0u;
}

__global__ void absmax_kernel(const float* __restrict__ Wf,
                              const float* __restrict__ Wc,
                              const float* __restrict__ Wg,
                              unsigned* __restrict__ amax) {
  const float* W = (blockIdx.y == 0) ? Wf : (blockIdx.y == 1) ? Wc : Wg;
  __shared__ float red[256];
  float m = 0.f;
  for (int i = blockIdx.x * 256 + threadIdx.x; i < EE * HH; i += gridDim.x * 256)
    m = fmaxf(m, fabsf(W[i]));
  red[threadIdx.x] = m; __syncthreads();
  for (int s = 128; s > 0; s >>= 1) {
    if (threadIdx.x < s) red[threadIdx.x] = fmaxf(red[threadIdx.x], red[threadIdx.x + s]);
    __syncthreads();
  }
  if (threadIdx.x == 0) atomicMax(&amax[blockIdx.y], __float_as_uint(red[0]));
}

__global__ void prep_weights_kernel(const float* __restrict__ Wf,
                                    const float* __restrict__ Wc,
                                    const float* __restrict__ Wg,
                                    const unsigned* __restrict__ amax,
                                    unsigned* __restrict__ wf_p,
                                    unsigned* __restrict__ wc_p,
                                    unsigned* __restrict__ wg_p,
                                    unsigned* __restrict__ wgT_p) {
  const int nt = blockIdx.x;
  const int which = blockIdx.y;
  if (which < 3) {
    const float* W = (which == 0) ? Wf : (which == 1) ? Wc : Wg;
    unsigned* dst  = (which == 0) ? wf_p : (which == 1) ? wc_p : wg_p;
    const float sc = 127.f / (__uint_as_float(amax[which]) + 1e-5f);
    for (int idx = threadIdx.x; idx < KT8 * 32 * 8; idx += blockDim.x) {
      const int kt = idx >> 8, lane = (idx >> 3) & 31, v = idx & 7;
      const int n = nt * 16 + (lane & 15);
      const int kb = kt * 64 + ((v >= 4) ? 32 : 0) + ((lane >= 16) ? 16 : 0) + (v & 3) * 4;
      unsigned dw = 0;
      #pragma unroll
      for (int b = 0; b < 4; ++b) {
        const float w  = W[(long)(kb + b) * HH + n];
        const float wq = rintf(fminf(fmaxf(w * sc, -128.f), 127.f));
        const int   t  = (wq > 0.f) - (wq < 0.f);
        dw |= ((unsigned)(t & 0xff)) << (8 * b);
      }
      dst[((nt * KT8 + kt) * 32 + lane) * 8 + v] = dw;
    }
  } else {
    for (int idx = threadIdx.x; idx < KT16 * 32 * 8; idx += blockDim.x) {
      const int kt = idx >> 8, lane = (idx >> 3) & 31, v = idx & 7;
      const int n  = nt * 16 + (lane & 15);
      const int kb = kt * 32 + (v >> 2) * 16 + ((lane >= 16) ? 8 : 0) + (v & 3) * 2;
      const unsigned lo = f2bf(Wg[(long)n * EE + kb]);
      const unsigned hi = f2bf(Wg[(long)n * EE + kb + 1]);
      wgT_p[((nt * KT16 + kt) * 32 + lane) * 8 + v] = lo | (hi << 16);
    }
  }
}

__global__ void prep_token_kernel(const float* __restrict__ x,
                                  const float* __restrict__ rms_scale,
                                  unsigned* __restrict__ xq_p,
                                  unsigned* __restrict__ xbf_p,
                                  float* __restrict__ inv_s) {
  const int tok = blockIdx.x;
  const float* xr = x + (long)tok * EE;
  __shared__ float xs[EE];
  __shared__ float xn[EE];
  __shared__ float red[256];
  __shared__ float sh_s;
  const int tid = threadIdx.x;

  float ss = 0.f;
  for (int i = tid; i < EE; i += 256) { float v = xr[i]; xs[i] = v; ss += v * v; }
  red[tid] = ss; __syncthreads();
  for (int s = 128; s > 0; s >>= 1) { if (tid < s) red[tid] += red[tid + s]; __syncthreads(); }
  const float rms = rsqrtf(red[0] / (float)EE + 1e-5f);
  __syncthreads();

  float mx = 0.f;
  for (int i = tid; i < EE; i += 256) {
    float v = rms_scale[i] * (xs[i] * rms); xn[i] = v; mx = fmaxf(mx, fabsf(v));
  }
  red[tid] = mx; __syncthreads();
  for (int s = 128; s > 0; s >>= 1) { if (tid < s) red[tid] = fmaxf(red[tid], red[tid + s]); __syncthreads(); }
  if (tid == 0) {
    float s = fminf(fmaxf(127.f / (red[0] + 1e-5f), 1e-3f), 1e3f);
    sh_s = s; inv_s[tok] = 1.f / s;
  }
  __syncthreads();
  const float s = sh_s;

  const int mt = tok >> 4, r = tok & 15;
  for (int idx = tid; idx < 512; idx += 256) {
    const int kt = idx >> 4, lh = (idx >> 3) & 1, v = idx & 7;
    const int lane = r + lh * 16;
    const int kb = kt * 64 + (v >> 2) * 32 + lh * 16 + (v & 3) * 4;
    unsigned dw = 0;
    #pragma unroll
    for (int b = 0; b < 4; ++b) {
      float q = fminf(fmaxf(rintf(s * xn[kb + b]), -128.f), 127.f);
      dw |= ((unsigned)((int)q & 0xff)) << (8 * b);
    }
    xq_p[((mt * KT8 + kt) * 32 + lane) * 8 + v] = dw;
  }
  for (int idx = tid; idx < 1024; idx += 256) {
    const int kt = idx >> 4, lh = (idx >> 3) & 1, v = idx & 7;
    const int lane = r + lh * 16;
    const int kb = kt * 32 + (v >> 2) * 16 + (v & 3) * 2 + lh * 8;
    const unsigned lo = f2bf(xs[kb]), hi = f2bf(xs[kb + 1]);
    xbf_p[((mt * KT16 + kt) * 32 + lane) * 8 + v] = lo | (hi << 16);
  }
}

__global__ void gemm3_int8_kernel(const unsigned* __restrict__ xq_p,
                                  const unsigned* __restrict__ wf_p,
                                  const unsigned* __restrict__ wc_p,
                                  const unsigned* __restrict__ wg_p,
                                  const float* __restrict__ inv_s,
                                  const float* __restrict__ bf,
                                  const float* __restrict__ bc,
                                  const float* __restrict__ bg,
                                  float* __restrict__ f_buf,
                                  float* __restrict__ c_buf,
                                  float* __restrict__ g_buf) {
  const int mt0  = blockIdx.x * 2;
  const int wave = threadIdx.x >> 5;
  const int lane = threadIdx.x & 31;
  const int nt   = blockIdx.y * 8 + wave;

  const v8i* A0 = (const v8i*)xq_p + (long)(mt0    ) * KT8 * 32 + lane;
  const v8i* A1 = (const v8i*)xq_p + (long)(mt0 + 1) * KT8 * 32 + lane;
  const v8i* Bf = (const v8i*)wf_p + (long)nt * KT8 * 32 + lane;
  const v8i* Bc = (const v8i*)wc_p + (long)nt * KT8 * 32 + lane;
  const v8i* Bg = (const v8i*)wg_p + (long)nt * KT8 * 32 + lane;

  v8i af0 = (v8i)0, af1 = (v8i)0;
  v8i ac0 = (v8i)0, ac1 = (v8i)0;
  v8i ag0 = (v8i)0, ag1 = (v8i)0;
  for (int kt = 0; kt < KT8; ++kt) {
    const v8i a0  = A0[kt * 32];
    const v8i a1  = A1[kt * 32];
    const v8i bfv = Bf[kt * 32];
    const v8i bcv = Bc[kt * 32];
    const v8i bgv = Bg[kt * 32];
    if (kt + 1 < KT8) {
      __builtin_prefetch(&Bf[(kt + 1) * 32], 0, 3);
      __builtin_prefetch(&Bc[(kt + 1) * 32], 0, 3);
      __builtin_prefetch(&Bg[(kt + 1) * 32], 0, 3);
    }
    af0 = __builtin_amdgcn_wmma_i32_16x16x64_iu8(true, a0, true, bfv, af0, false, false);
    af1 = __builtin_amdgcn_wmma_i32_16x16x64_iu8(true, a1, true, bfv, af1, false, false);
    ac0 = __builtin_amdgcn_wmma_i32_16x16x64_iu8(true, a0, true, bcv, ac0, false, false);
    ac1 = __builtin_amdgcn_wmma_i32_16x16x64_iu8(true, a1, true, bcv, ac1, false, false);
    ag0 = __builtin_amdgcn_wmma_i32_16x16x64_iu8(true, a0, true, bgv, ag0, false, false);
    ag1 = __builtin_amdgcn_wmma_i32_16x16x64_iu8(true, a1, true, bgv, ag1, false, false);
  }

  const int n    = nt * 16 + (lane & 15);
  const int mofs = (lane >= 16) ? 8 : 0;
  const float bfn = bf[n], bcn = bc[n], bgn = bg[n];
  #pragma unroll
  for (int r = 0; r < 8; ++r) {
    {
      const int tok  = mt0 * 16 + r + mofs;
      const float iv = inv_s[tok];
      const long  o  = (long)tok * HH + n;
      const float pf = (float)af0[r] * iv + bfn;
      const float pc = (float)ac0[r] * iv + bcn;
      const float pg = (float)ag0[r] * iv + bgn;
      f_buf[o] = sigm(pf);
      c_buf[o] = pc * sigm(pc);
      g_buf[o] = sigm(pg);
    }
    {
      const int tok  = (mt0 + 1) * 16 + r + mofs;
      const float iv = inv_s[tok];
      const long  o  = (long)tok * HH + n;
      const float pf = (float)af1[r] * iv + bfn;
      const float pc = (float)ac1[r] * iv + bcn;
      const float pg = (float)ag1[r] * iv + bgn;
      f_buf[o] = sigm(pf);
      c_buf[o] = pc * sigm(pc);
      g_buf[o] = sigm(pg);
    }
  }
}

__global__ void gemm_cg_bf16_kernel(const unsigned* __restrict__ xbf_p,
                                    const unsigned* __restrict__ wgT_p,
                                    float* __restrict__ cg_buf) {
  const int mt0  = blockIdx.x * 2;
  const int wave = threadIdx.x >> 5;
  const int lane = threadIdx.x & 31;
  const int nt   = blockIdx.y * 8 + wave;

  const v16bf* A0 = (const v16bf*)xbf_p + (long)(mt0    ) * KT16 * 32 + lane;
  const v16bf* A1 = (const v16bf*)xbf_p + (long)(mt0 + 1) * KT16 * 32 + lane;
  const v16bf* B  = (const v16bf*)wgT_p + (long)nt * KT16 * 32 + lane;

  v8f acc0 = (v8f)0.f, acc1 = (v8f)0.f;
  for (int kt = 0; kt < KT16; ++kt) {
    const v16bf a0 = A0[kt * 32];
    const v16bf a1 = A1[kt * 32];
    const v16bf b  = B[kt * 32];
    if (kt + 1 < KT16) __builtin_prefetch(&B[(kt + 1) * 32], 0, 3);
    acc0 = __builtin_amdgcn_wmma_f32_16x16x32_bf16(false, a0, false, b,
                                                   (short)0, acc0, false, false);
    acc1 = __builtin_amdgcn_wmma_f32_16x16x32_bf16(false, a1, false, b,
                                                   (short)0, acc1, false, false);
  }

  const int n    = nt * 16 + (lane & 15);
  const int mofs = (lane >= 16) ? 8 : 0;
  #pragma unroll
  for (int r = 0; r < 8; ++r) {
    const int tok0 = mt0 * 16 + r + mofs;
    const int tok1 = (mt0 + 1) * 16 + r + mofs;
    cg_buf[(long)tok0 * HH + n] = sigm(acc0[r]);
    cg_buf[(long)tok1 * HH + n] = sigm(acc1[r]);
  }
}

__global__ void scan_kernel(const float* __restrict__ x,
                            const float* __restrict__ f_buf,
                            const float* __restrict__ c_buf,
                            const float* __restrict__ g_buf,
                            const float* __restrict__ cg_buf,
                            float* __restrict__ out) {
  const int idx = blockIdx.x * blockDim.x + threadIdx.x;
  const int b = idx >> 11;
  const int j = idx & (HH - 1);
  float h = 0.f;
  for (int t = 0; t < SS; ++t) {
    const long o = ((long)(b * SS + t)) * HH + j;
    const float f = f_buf[o], c = c_buf[o], g = g_buf[o], cg = cg_buf[o], xv = x[o];
    const float hn = f * h + (1.f - f) * c;
    out[o] = g * hn;
    h = cg * xv + (1.f - cg) * hn;
  }
}

extern "C" void kernel_launch(void* const* d_in, const int* in_sizes, int n_in,
                              void* d_out, int out_size, void* d_ws, size_t ws_size,
                              hipStream_t stream) {
  (void)in_sizes; (void)n_in; (void)out_size; (void)ws_size;
  const float* x  = (const float*)d_in[0];
  const float* rs = (const float*)d_in[1];
  const float* Wf = (const float*)d_in[2];
  const float* Wc = (const float*)d_in[3];
  const float* Wg = (const float*)d_in[4];
  const float* bf = (const float*)d_in[5];
  const float* bc = (const float*)d_in[6];
  const float* bg = (const float*)d_in[7];

  char* ws = (char*)d_ws;
  size_t off = 0;
  auto alloc = [&](size_t bytes) -> char* {
    char* p = ws + off;
    off = (off + bytes + 255) & ~(size_t)255;
    return p;
  };
  unsigned* amax   = (unsigned*)alloc(4 * sizeof(unsigned));
  float*    inv_s  = (float*)   alloc((size_t)NTOK * 4);
  unsigned* xq_p   = (unsigned*)alloc((size_t)NTOK * EE);
  unsigned* xbf_p  = (unsigned*)alloc((size_t)NTOK * EE * 2);
  unsigned* wf_p   = (unsigned*)alloc((size_t)EE * HH);
  unsigned* wc_p   = (unsigned*)alloc((size_t)EE * HH);
  unsigned* wg_p   = (unsigned*)alloc((size_t)EE * HH);
  unsigned* wgT_p  = (unsigned*)alloc((size_t)EE * HH * 2);
  float*    f_buf  = (float*)   alloc((size_t)NTOK * HH * 4);
  float*    c_buf  = (float*)   alloc((size_t)NTOK * HH * 4);
  float*    g_buf  = (float*)   alloc((size_t)NTOK * HH * 4);
  float*    cg_buf = (float*)   alloc((size_t)NTOK * HH * 4);
  float*    out    = (float*)d_out;

  init_amax_kernel<<<1, 32, 0, stream>>>(amax);
  absmax_kernel<<<dim3(512, 3), 256, 0, stream>>>(Wf, Wc, Wg, amax);
  prep_weights_kernel<<<dim3(NT, 4), 256, 0, stream>>>(Wf, Wc, Wg, amax,
                                                       wf_p, wc_p, wg_p, wgT_p);
  prep_token_kernel<<<NTOK, 256, 0, stream>>>(x, rs, xq_p, xbf_p, inv_s);
  gemm3_int8_kernel<<<dim3(MT / 2, NT / 8), 256, 0, stream>>>(xq_p, wf_p, wc_p, wg_p,
                                                              inv_s, bf, bc, bg,
                                                              f_buf, c_buf, g_buf);
  gemm_cg_bf16_kernel<<<dim3(MT / 2, NT / 8), 256, 0, stream>>>(xbf_p, wgT_p, cg_buf);
  scan_kernel<<<(BB * HH) / 256, 256, 0, stream>>>(x, f_buf, c_buf, g_buf, cg_buf, out);
}
